// SimpleNet_10909216932634
// MI455X (gfx1250) — matrix operand order repaired and checked by reading, not compiled, not run on hardware
//
#include <hip/hip_runtime.h>
#include <hip/hip_bf16.h>
#include <math.h>

// ---------------------------------------------------------------------------
// CDNA5 (gfx1250) implementation of the bipartite GNN forward pass.
// Dense math via v_wmma_f32_16x16x32_f16 (wave32, 16-row tiles); hot weights
// staged in LDS (ds_read_b128 B-fragments), persistent waves, s_wait_dscnt for
// intra-wave LDS transposes.
// ---------------------------------------------------------------------------

typedef __attribute__((ext_vector_type(16))) _Float16 v16h;
typedef __attribute__((ext_vector_type(8)))  _Float16 v8h;
typedef __attribute__((ext_vector_type(8)))  float    v8f;

#define DEVI __device__ __forceinline__

union V16U { v16h v; v8h h[2]; };

DEVI v8f wmma_f16(v16h a, v16h b, v8f c) {
  return __builtin_amdgcn_wmma_f32_16x16x32_f16(false, a, false, b, (short)0, c,
                                                false, false);
}

// B fragment from a 64x64 f16 weight block (row n = output feature, contiguous
// K). Lane L: column n = ntile*16 + (L&15), K (L>>4)*8 .. +7 and +16 in window.
// Works for both global and LDS pointers (generic addressing).
DEVI v16h load_b64(const _Float16* Wt, int K, int ntile, int kc) {
  int lane = threadIdx.x & 31;
  int n  = (ntile << 4) + (lane & 15);
  int k0 = (kc << 5) + ((lane >> 4) << 3);
  const _Float16* p = Wt + (size_t)n * K + k0;
  V16U u;
  u.h[0] = *(const v8h*)p;
  u.h[1] = *(const v8h*)(p + 16);
  return u.v;
}

// A fragment from f32 row pointer (row resolved per lane: m = lane&15).
// Halves 0..7 = K (lane>>4)*8 .. +7 ; halves 8..15 = +16.
DEVI v16h a_from_f32(const float* __restrict__ rowp, int kc) {
  int lane = threadIdx.x & 31;
  int k0 = (kc << 5) + ((lane >> 4) << 3);
  v16h a;
#pragma unroll
  for (int j = 0; j < 8; ++j) a[j] = (_Float16)rowp[k0 + j];
#pragma unroll
  for (int j = 0; j < 8; ++j) a[8 + j] = (_Float16)rowp[k0 + 16 + j];
  return a;
}

DEVI v16h a_from_f16(const _Float16* rowp, int kc) {
  int lane = threadIdx.x & 31;
  int k0 = (kc << 5) + ((lane >> 4) << 3);
  V16U u;
  u.h[0] = *(const v8h*)(rowp + k0);
  u.h[1] = *(const v8h*)(rowp + k0 + 16);
  return u.v;
}

// Cooperative global->LDS copy of an 8 KB (64x64 f16) weight block.
DEVI void stage_wt(_Float16* dst, const _Float16* src, int tid, int nthr) {
  const uint4* s = (const uint4*)src;
  uint4* d = (uint4*)dst;
#pragma unroll 1
  for (int i = tid; i < 512; i += nthr) d[i] = s[i];  // 512 * 16B = 8KB
}

DEVI unsigned f2ord(float x) {
  unsigned u = __float_as_uint(x);
  return (u & 0x80000000u) ? ~u : (u | 0x80000000u);
}
DEVI float ord2f(unsigned u) {
  return (u & 0x80000000u) ? __uint_as_float(u & 0x7fffffffu)
                           : __uint_as_float(~u);
}

// Wave-local LDS RAW fence: DS ops of a wave are in-order; lanes are lockstep,
// so waiting DScnt==0 makes same-wave cross-lane LDS writes visible.
DEVI void wave_lds_fence() { asm volatile("s_wait_dscnt 0x0" ::: "memory"); }

// ---------------------------------------------------------------------------
// Weight convert: W (rows x 64 f32, row-major) -> Wt f16, Wt[n*K+k]=W[rowOff+k][n]
__global__ void conv_wt_kernel(const float* __restrict__ W, int rowOff, int K,
                               _Float16* __restrict__ dst) {
  int i = blockIdx.x * blockDim.x + threadIdx.x;
  if (i >= 64 * K) return;
  int n = i / K, k = i % K;
  dst[i] = (_Float16)W[(size_t)(rowOff + k) * 64 + n];
}

// BN finish: s = rsqrt(var+eps)*g ; t = beta - mean*s
__global__ void bn_finish_kernel(const float* __restrict__ sum,
                                 const float* __restrict__ sq, float n,
                                 const float* __restrict__ g,
                                 const float* __restrict__ beta,
                                 float* __restrict__ s, float* __restrict__ t) {
  int f = threadIdx.x;
  float m = sum[f] / n;
  float v = sq[f] / n - m * m;
  float sc = rsqrtf(v + 1e-5f) * g[f];
  s[f] = sc;
  t[f] = beta[f] - m * sc;
}

// Fold BN(s,t) of a 64-dim feature into the consuming 64x64 weight block
// (rows rowOff..rowOff+63 of W1) and into the layer-1 bias.
__global__ void fold_ee_kernel(const float* __restrict__ W1, int rowOff,
                               const float* __restrict__ s,
                               const float* __restrict__ t,
                               const float* __restrict__ b1,
                               _Float16* __restrict__ WtD,
                               float* __restrict__ bias1) {
  int n = threadIdx.x;  // 64 threads
  float acc = b1[n];
  for (int k = 0; k < 64; ++k) {
    float wkn = W1[(size_t)(rowOff + k) * 64 + n];
    WtD[n * 64 + k] = (_Float16)(wkn * s[k]);
    acc += t[k] * wkn;
  }
  bias1[n] = acc;
}

// ---------------------------------------------------------------------------
// Generic: Y[rows x 64] = act(X[:, colx:colx+K] @ W(Kx64) + b), WMMA.
__global__ void __launch_bounds__(256)
mm_rows_kernel(int nRows, const float* __restrict__ X, int ldx, int colx, int K,
               const _Float16* __restrict__ Wt, const float* __restrict__ bias,
               int relu, float* __restrict__ Y, int ldy, int coly) {
  int wave = threadIdx.x >> 5, lane = threadIdx.x & 31;
  int tile0 = (blockIdx.x * 8 + wave) * 16;
  if (tile0 >= nRows) return;
  int row = tile0 + (lane & 15);
  const float* xp = X + (size_t)row * ldx + colx;
  v8f c[4] = {};
  for (int kc = 0; kc < K / 32; ++kc) {
    v16h a = a_from_f32(xp, kc);
#pragma unroll
    for (int nt = 0; nt < 4; ++nt) c[nt] = wmma_f16(a, load_b64(Wt, K, nt, kc), c[nt]);
  }
#pragma unroll
  for (int nt = 0; nt < 4; ++nt) {
    int f = (lane & 15) + (nt << 4);
    float bb = bias[f];
#pragma unroll
    for (int r = 0; r < 8; ++r) {
      int mm = ((lane >> 4) << 3) + r;
      float v = c[nt][r] + bb;
      if (relu) v = fmaxf(v, 0.f);
      Y[(size_t)(tile0 + mm) * ldy + coly + f] = v;
    }
  }
}

// ---------------------------------------------------------------------------
// MLP with tiny input dim (1 or 2 scalars per row): u = relu(x*w1 + b1) built
// directly in the WMMA A-fragment registers, then u @ W2(64x64)+b2.
// W2 staged in LDS once per block; persistent waves grid-stride over tiles.
template <int DIN, bool MINUS, bool F16OUT>
__global__ void __launch_bounds__(256)
smallin_kernel(int nRows, const float* __restrict__ Xs,
               const float* __restrict__ Xm, const float* __restrict__ w1,
               const float* __restrict__ b1, const _Float16* __restrict__ Wt2,
               const float* __restrict__ b2, int relu2, float* __restrict__ Yf,
               _Float16* __restrict__ Yh, int ldy, int coly,
               float* __restrict__ statSum, float* __restrict__ statSq) {
  __shared__ __align__(16) _Float16 ldsW[4096];
  stage_wt(ldsW, Wt2, threadIdx.x, 256);
  __syncthreads();
  int wave = threadIdx.x >> 5, lane = threadIdx.x & 31;
  int nT = (nRows + 15) >> 4;
  for (int tile = blockIdx.x * 8 + wave; tile < nT; tile += gridDim.x * 8) {
    int tile0 = tile << 4;
    int row = tile0 + (lane & 15);
    float x0 = Xs[(size_t)row * DIN];
    float x1 = (DIN == 2) ? Xs[(size_t)row * DIN + 1] : 0.f;
    if (MINUS) x0 -= Xm[row];
    v8f c[4] = {};
    for (int kc = 0; kc < 2; ++kc) {
      int k0 = (kc << 5) + ((lane >> 4) << 3);
      v16h a;
#pragma unroll
      for (int j = 0; j < 16; ++j) {
        int k = k0 + ((j < 8) ? j : (j + 8));  // halves 8..15 -> +16 offset
        float u = x0 * w1[k] + b1[k];
        if (DIN == 2) u += x1 * w1[64 + k];
        a[j] = (_Float16)fmaxf(u, 0.f);
      }
#pragma unroll
      for (int nt = 0; nt < 4; ++nt)
        c[nt] = wmma_f16(a, load_b64(ldsW, 64, nt, kc), c[nt]);
    }
#pragma unroll
    for (int nt = 0; nt < 4; ++nt) {
      int f = (lane & 15) + (nt << 4);
      float bb = b2[f];
      float ss = 0.f, qq = 0.f;
#pragma unroll
      for (int r = 0; r < 8; ++r) {
        int mm = ((lane >> 4) << 3) + r;
        float v = c[nt][r] + bb;
        if (relu2) v = fmaxf(v, 0.f);
        size_t ro = (size_t)(tile0 + mm);
        if (F16OUT) Yh[ro * 64 + f] = (_Float16)v;
        else        Yf[ro * ldy + coly + f] = v;
        ss += v; qq += v * v;
      }
      if (statSum) {
        ss += __shfl_xor(ss, 16);
        qq += __shfl_xor(qq, 16);
        if (lane < 16) { atomicAdd(&statSum[f], ss); atomicAdd(&statSq[f], qq); }
      }
    }
  }
}

// ---------------------------------------------------------------------------
// Fused edge-message kernel: z1 = relu( X0[idx0]@WtA + X1[idx1]@WtB
//   (+ X2[idx2]@WtC) (+ a[idxS] (x) w1c) + EE@WtE + bias1 )   [BN of EE folded]
// z2 = relu(z1 @ Wt2 + b2) ; scatter-add z2 into agg[idxT] ; BN stats of z2.
// All 5 weight blocks live in LDS; persistent waves; per-tile transpose uses a
// wave-local s_wait_dscnt fence (no block barrier in the loop).
__global__ void __launch_bounds__(128)
edge_msg_kernel(int nE,
                const int* __restrict__ idx0, const float* __restrict__ X0,
                int ld0, int col0, const _Float16* __restrict__ WtA,
                const int* __restrict__ idx1, const float* __restrict__ X1,
                int ld1, int col1, const _Float16* __restrict__ WtB,
                const int* __restrict__ idx2, const float* __restrict__ X2,
                const _Float16* __restrict__ WtC, const int* __restrict__ idxS,
                const float* __restrict__ avec, const float* __restrict__ w1c,
                const _Float16* __restrict__ EE, const _Float16* __restrict__ WtE,
                const float* __restrict__ bias1, const _Float16* __restrict__ Wt2,
                const float* __restrict__ b2, const int* __restrict__ idxT,
                float* __restrict__ agg, float* __restrict__ statSum,
                float* __restrict__ statSq) {
  // LDS: 5 weight blocks (5*4096 halves = 40KB) + 4 wave transpose buffers (8KB)
  __shared__ __align__(16) _Float16 lds[5 * 4096 + 4 * 1024];
  stage_wt(lds + 0 * 4096, WtA, threadIdx.x, 128);
  stage_wt(lds + 1 * 4096, WtB, threadIdx.x, 128);
  if (idx2) stage_wt(lds + 2 * 4096, WtC, threadIdx.x, 128);
  stage_wt(lds + 3 * 4096, WtE, threadIdx.x, 128);
  stage_wt(lds + 4 * 4096, Wt2, threadIdx.x, 128);
  __syncthreads();

  int wave = threadIdx.x >> 5, lane = threadIdx.x & 31;
  _Float16* ldsZ = lds + 5 * 4096 + wave * 1024;
  int nT = nE >> 4;
  for (int tile = blockIdx.x * 4 + wave; tile < nT; tile += gridDim.x * 4) {
    int tile0 = tile << 4;
    int rowe = tile0 + (lane & 15);  // edge index for A-fragment loads
    int r0 = idx0[rowe];
    int r1 = idx1[rowe];
    int r2 = idx2 ? idx2[rowe] : 0;

    v8f c[4] = {};
    for (int kc = 0; kc < 2; ++kc) {
      v16h a = a_from_f32(X0 + (size_t)r0 * ld0 + col0, kc);
#pragma unroll
      for (int nt = 0; nt < 4; ++nt)
        c[nt] = wmma_f16(a, load_b64(lds + 0 * 4096, 64, nt, kc), c[nt]);
    }
    for (int kc = 0; kc < 2; ++kc) {
      v16h a = a_from_f32(X1 + (size_t)r1 * ld1 + col1, kc);
#pragma unroll
      for (int nt = 0; nt < 4; ++nt)
        c[nt] = wmma_f16(a, load_b64(lds + 1 * 4096, 64, nt, kc), c[nt]);
    }
    if (idx2) {
      for (int kc = 0; kc < 2; ++kc) {
        v16h a = a_from_f32(X2 + (size_t)r2 * 64, kc);
#pragma unroll
        for (int nt = 0; nt < 4; ++nt)
          c[nt] = wmma_f16(a, load_b64(lds + 2 * 4096, 64, nt, kc), c[nt]);
      }
    }
    for (int kc = 0; kc < 2; ++kc) {
      v16h a = a_from_f16(EE + (size_t)rowe * 64, kc);
#pragma unroll
      for (int nt = 0; nt < 4; ++nt)
        c[nt] = wmma_f16(a, load_b64(lds + 3 * 4096, 64, nt, kc), c[nt]);
    }

    // per-D-row scatter target + optional rank-1 scalar term
    int tIdx[8]; float av[8];
#pragma unroll
    for (int r = 0; r < 8; ++r) {
      int e2 = tile0 + ((lane >> 4) << 3) + r;
      tIdx[r] = idxT[e2];
      av[r] = idxS ? avec[idxS[e2]] : 0.f;
    }

    // finish layer-1, relu, stage to this wave's LDS region (D -> row-major)
#pragma unroll
    for (int nt = 0; nt < 4; ++nt) {
      int f = (lane & 15) + (nt << 4);
      float bb = bias1[f];
      float wc = idxS ? w1c[f] : 0.f;
#pragma unroll
      for (int r = 0; r < 8; ++r) {
        int mm = ((lane >> 4) << 3) + r;
        float v = fmaxf(c[nt][r] + bb + av[r] * wc, 0.f);
        ldsZ[mm * 64 + f] = (_Float16)v;
      }
    }
    wave_lds_fence();  // s_wait_dscnt 0: wave-local LDS RAW

    v8f d[4] = {};
    const _Float16* lrow = ldsZ + (lane & 15) * 64;
    for (int kc = 0; kc < 2; ++kc) {
      v16h a = a_from_f16(lrow, kc);
#pragma unroll
      for (int nt = 0; nt < 4; ++nt)
        d[nt] = wmma_f16(a, load_b64(lds + 4 * 4096, 64, nt, kc), d[nt]);
    }
#pragma unroll
    for (int nt = 0; nt < 4; ++nt) {
      int f = (lane & 15) + (nt << 4);
      float bb = b2[f];
      float ss = 0.f, qq = 0.f;
#pragma unroll
      for (int r = 0; r < 8; ++r) {
        float v = fmaxf(d[nt][r] + bb, 0.f);
        ss += v; qq += v * v;
        atomicAdd(&agg[(size_t)tIdx[r] * 64 + f], v);
      }
      ss += __shfl_xor(ss, 16);
      qq += __shfl_xor(qq, 16);
      if (lane < 16) { atomicAdd(&statSum[f], ss); atomicAdd(&statSq[f], qq); }
    }
    wave_lds_fence();  // protect ldsZ reuse next iteration (WAR)
  }
}

// ---------------------------------------------------------------------------
// Small helpers
__global__ void dot_head_kernel(int n, const float* __restrict__ H,
                                const float* __restrict__ w,
                                const float* __restrict__ bptr, int sig,
                                float* __restrict__ out) {
  int i = blockIdx.x * blockDim.x + threadIdx.x;
  if (i >= n) return;
  float s = bptr[0];
  const float* h = H + (size_t)i * 64;
#pragma unroll
  for (int k = 0; k < 64; ++k) s += h[k] * w[k];
  out[i] = sig ? 1.f / (1.f + expf(-s)) : s;
}

__global__ void count_kernel(int n, const int* __restrict__ idx,
                             float* __restrict__ cnt) {
  int i = blockIdx.x * blockDim.x + threadIdx.x;
  if (i < n) atomicAdd(&cnt[idx[i]], 1.f);
}

__global__ void tmp_kernel(int n, const int* __restrict__ src,
                           const int* __restrict__ tgt,
                           const float* __restrict__ a,
                           const float* __restrict__ ef,
                           float* __restrict__ tmp) {
  int i = blockIdx.x * blockDim.x + threadIdx.x;
  if (i < n) atomicAdd(&tmp[tgt[i]], a[src[i]] * ef[i]);
}

__global__ void seg_bnmax_kernel(int n, float* __restrict__ Z,
                                 const float* __restrict__ s,
                                 const float* __restrict__ t,
                                 const int* __restrict__ gidx,
                                 unsigned* __restrict__ gmax) {
  int i = blockIdx.x * blockDim.x + threadIdx.x;
  if (i >= n * 64) return;
  int c = i >> 6, f = i & 63;
  float z = Z[i] * s[f] + t[f];
  Z[i] = z;
  atomicMax(&gmax[gidx[c] * 64 + f], f2ord(z));
}

__global__ void seg_exp_kernel(int n, float* __restrict__ Z,
                               const int* __restrict__ gidx,
                               const unsigned* __restrict__ gmax,
                               float* __restrict__ den) {
  int i = blockIdx.x * blockDim.x + threadIdx.x;
  if (i >= n * 64) return;
  int c = i >> 6, f = i & 63;
  float ex = expf(Z[i] - ord2f(gmax[gidx[c] * 64 + f]));
  Z[i] = ex;
  atomicAdd(&den[gidx[c] * 64 + f], ex);
}

__global__ void seg_div_kernel(int n, float* __restrict__ Z,
                               const int* __restrict__ gidx,
                               const float* __restrict__ den) {
  int i = blockIdx.x * blockDim.x + threadIdx.x;
  if (i >= n * 64) return;
  int c = i >> 6, f = i & 63;
  Z[i] = Z[i] / (den[gidx[c] * 64 + f] + 1e-16f);
}

// xc_new = relu( s*agg/max(cnt,1) + t*(cnt>0) )   (BN folded through mean-agg)
__global__ void agg_finish_kernel(int n, const float* __restrict__ agg,
                                  const float* __restrict__ cnt,
                                  const float* __restrict__ s,
                                  const float* __restrict__ t,
                                  float* __restrict__ Y, int ldy, int coly) {
  int i = blockIdx.x * blockDim.x + threadIdx.x;
  if (i >= n * 64) return;
  int node = i >> 6, f = i & 63;
  float c = cnt[node];
  float v = s[f] * agg[i] / fmaxf(c, 1.f) + (c > 0.f ? t[f] : 0.f);
  Y[(size_t)node * ldy + coly + f] = fmaxf(v, 0.f);
}

// ---------------------------------------------------------------------------
extern "C" void kernel_launch(void* const* d_in, const int* in_sizes, int n_in,
                              void* d_out, int out_size, void* d_ws,
                              size_t ws_size, hipStream_t stream) {
  const float* VNF = (const float*)d_in[0];
  const float* CNF = (const float*)d_in[1];
  const float* EFV = (const float*)d_in[2];
  const float* EFC = (const float*)d_in[3];
  const float* RHS = (const float*)d_in[4];
  const float* P   = (const float*)d_in[5];
  const int*   EIV = (const int*)d_in[6];
  const int*   EIC = (const int*)d_in[7];
  const int*   GIDX = (const int*)d_in[8];
  const int V = in_sizes[0] / 2, C = in_sizes[1] / 2, E = in_sizes[2];
  const int* src_v = EIV;       const int* tgt_c = EIV + E;
  const int* src_c = EIC;       const int* tgt_v = EIC + E;

  // ---- params flat offsets (JAX pytree order: dict keys sorted) ----
  long o = 0;
  auto tk = [&](long n) { long r = o; o += n; return r; };
  long ce_l1b = tk(64), ce_l1w = tk(128), ce_l2b = tk(64), ce_l2w = tk(4096);
  struct LP {
    long ass_l1b, ass_l1w, ass_l2b, ass_l2w;
    long cee_beta, cee_g, cee_l1b, cee_l1w, cee_l2b, cee_l2w;
    long cnn_beta, cnn_g, cnn_l1b, cnn_l1w, cnn_l2b, cnn_l2w;
    long err_beta, err_g, err_l1b, err_l1w, err_l2b, err_l2w;
    long vee_beta, vee_g, vee_l1b, vee_l1w, vee_l2b, vee_l2w;
    long vnn_beta, vnn_g, vnn_l1b, vnn_l1w, vnn_l2b, vnn_l2w;
  } L[2];
  for (int l = 0; l < 2; ++l) {
    L[l].ass_l1b = tk(64); L[l].ass_l1w = tk(4096); L[l].ass_l2b = tk(1); L[l].ass_l2w = tk(64);
    L[l].cee_beta = tk(64); L[l].cee_g = tk(64); L[l].cee_l1b = tk(64); L[l].cee_l1w = tk(64); L[l].cee_l2b = tk(64); L[l].cee_l2w = tk(4096);
    L[l].cnn_beta = tk(64); L[l].cnn_g = tk(64); L[l].cnn_l1b = tk(64); L[l].cnn_l1w = tk(256 * 64); L[l].cnn_l2b = tk(64); L[l].cnn_l2w = tk(4096);
    L[l].err_beta = tk(64); L[l].err_g = tk(64); L[l].err_l1b = tk(64); L[l].err_l1w = tk(64); L[l].err_l2b = tk(64); L[l].err_l2w = tk(4096);
    L[l].vee_beta = tk(64); L[l].vee_g = tk(64); L[l].vee_l1b = tk(64); L[l].vee_l1w = tk(64); L[l].vee_l2b = tk(64); L[l].vee_l2w = tk(4096);
    L[l].vnn_beta = tk(64); L[l].vnn_g = tk(64); L[l].vnn_l1b = tk(64); L[l].vnn_l1w = tk(193 * 64); L[l].vnn_l2b = tk(64); L[l].vnn_l2w = tk(4096);
  }
  long l1b = tk(64), l1w = tk(192 * 64), l2b = tk(64), l2w = tk(4096);
  long l3b = tk(64), l3w = tk(4096), l4b = tk(1), l4w = tk(64);
  long ve_l1b = tk(64), ve_l1w = tk(128), ve_l2b = tk(64), ve_l2w = tk(4096);

  // ---- workspace carve (~170 MB) ----
  char* wp = (char*)d_ws;
  auto carve = [&](size_t bytes) {
    void* p = (void*)wp;
    wp += (bytes + 255) & ~(size_t)255;
    return p;
  };
  float* XV   = (float*)carve((size_t)V * 192 * 4);   // x_var0|1|2 columns
  float* XC0  = (float*)carve((size_t)C * 64 * 4);
  float* XC1  = (float*)carve((size_t)C * 64 * 4);
  float* ERR  = (float*)carve((size_t)C * 64 * 4);
  float* AGG  = (float*)carve((size_t)((V > C) ? V : C) * 64 * 4);
  _Float16* EE = (_Float16*)carve((size_t)E * 64 * 2);
  float* H1   = (float*)carve((size_t)V * 64 * 4);
  float* H2   = (float*)carve((size_t)V * 64 * 4);
  float* AV   = (float*)carve((size_t)V * 4);
  float* TMP  = (float*)carve((size_t)C * 4);
  float* CNTC = (float*)carve((size_t)C * 4);
  float* CNTV = (float*)carve((size_t)V * 4);
  float* SSUM = (float*)carve(64 * 4);
  float* SSQ  = (float*)carve(64 * 4);
  float* SBN  = (float*)carve(64 * 4);
  float* TBN  = (float*)carve(64 * 4);
  float* BIAS1 = (float*)carve(64 * 4);
  unsigned* GMAX = (unsigned*)carve(32 * 64 * 4);
  float* DEN  = (float*)carve(32 * 64 * 4);
  auto carveH = [&]() { return (_Float16*)carve(64 * 64 * 2); };
  _Float16* WtVE2 = carveH();
  _Float16* WtCE2 = carveH();
  _Float16 *WtASS1[2], *WtERR2[2], *WtVEE2[2], *WtCEE2[2];
  _Float16 *WtVA[2], *WtVB[2], *WtV2[2], *WtCA[2], *WtCB[2], *WtCC[2], *WtC2[2];
  for (int l = 0; l < 2; ++l) {
    WtASS1[l] = carveH(); WtERR2[l] = carveH(); WtVEE2[l] = carveH(); WtCEE2[l] = carveH();
    WtVA[l] = carveH(); WtVB[l] = carveH(); WtV2[l] = carveH();
    WtCA[l] = carveH(); WtCB[l] = carveH(); WtCC[l] = carveH(); WtC2[l] = carveH();
  }
  _Float16* WtL1 = (_Float16*)carve((size_t)64 * 192 * 2);
  _Float16* WtL2 = carveH();
  _Float16* WtL3 = carveH();
  _Float16* WtD  = carveH();  // folded ee-block, reused sequentially

  auto conv = [&](long woff, int rowOff, int K, _Float16* dst) {
    int n = 64 * K;
    conv_wt_kernel<<<(n + 255) / 256, 256, 0, stream>>>(P + woff, rowOff, K, dst);
  };
  auto zero = [&](void* p, size_t b) { hipMemsetAsync(p, 0, b, stream); };
  const int gV = ((V + 15) / 16 + 7) / 8;   // direct grids for V/C row passes
  const int gC = ((C + 15) / 16 + 7) / 8;
  const int gEsm = 1024;                    // persistent grid, E-row smallin
  const int gMsg = 1250;                    // persistent grid, edge messages
  const int gEW = (E + 255) / 256;
  const int gC64 = (C * 64 + 255) / 256;
  const int gV64 = (V * 64 + 255) / 256;

  // ---- pre: counts + weight conversion + encoders ----
  zero(CNTC, (size_t)C * 4); zero(CNTV, (size_t)V * 4);
  count_kernel<<<gEW, 256, 0, stream>>>(E, tgt_c, CNTC);
  count_kernel<<<gEW, 256, 0, stream>>>(E, tgt_v, CNTV);
  conv(ve_l2w, 0, 64, WtVE2); conv(ce_l2w, 0, 64, WtCE2);
  for (int l = 0; l < 2; ++l) {
    conv(L[l].ass_l1w, 0, 64, WtASS1[l]);
    conv(L[l].err_l2w, 0, 64, WtERR2[l]);
    conv(L[l].vee_l2w, 0, 64, WtVEE2[l]);
    conv(L[l].cee_l2w, 0, 64, WtCEE2[l]);
    conv(L[l].vnn_l1w, 0, 64, WtVA[l]);   // x_con block
    conv(L[l].vnn_l1w, 64, 64, WtVB[l]);  // x_var block
    conv(L[l].vnn_l2w, 0, 64, WtV2[l]);
    conv(L[l].cnn_l1w, 0, 64, WtCA[l]);   // x_var block
    conv(L[l].cnn_l1w, 64, 64, WtCB[l]);  // xc_new block
    conv(L[l].cnn_l1w, 128, 64, WtCC[l]); // err block
    conv(L[l].cnn_l2w, 0, 64, WtC2[l]);
  }
  conv(l1w, 0, 192, WtL1); conv(l2w, 0, 64, WtL2); conv(l3w, 0, 64, WtL3);

  smallin_kernel<2, false, false><<<gV, 256, 0, stream>>>(
      V, VNF, nullptr, P + ve_l1w, P + ve_l1b, WtVE2, P + ve_l2b, 0,
      XV, nullptr, 192, 0, nullptr, nullptr);
  smallin_kernel<2, false, false><<<gC, 256, 0, stream>>>(
      C, CNF, nullptr, P + ce_l1w, P + ce_l1b, WtCE2, P + ce_l2b, 0,
      XC0, nullptr, 64, 0, nullptr, nullptr);

  // ---- layers ----
  for (int l = 0; l < 2; ++l) {
    const LP& q = L[l];
    float* xc_cur = (l == 0) ? XC0 : XC1;
    float* xc_new = (l == 0) ? XC1 : XC0;
    int colv = 64 * l;

    // a = sigmoid(mlp2(x_var))
    mm_rows_kernel<<<gV, 256, 0, stream>>>(V, XV, 192, colv, 64, WtASS1[l],
                                           P + q.ass_l1b, 1, H1, 64, 0);
    dot_head_kernel<<<(V + 255) / 256, 256, 0, stream>>>(
        V, H1, P + q.ass_l2w, P + q.ass_l2b, 1, AV);

    // tmp = segsum(a[src_v]*e_var -> tgt_c)
    zero(TMP, (size_t)C * 4);
    tmp_kernel<<<gEW, 256, 0, stream>>>(E, src_v, tgt_c, AV, EFV, TMP);

    // err = seg_softmax(BN(mlp(tmp - rhs)))
    zero(SSUM, 256); zero(SSQ, 256);
    smallin_kernel<1, true, false><<<gC, 256, 0, stream>>>(
        C, TMP, RHS, P + q.err_l1w, P + q.err_l1b, WtERR2[l], P + q.err_l2b, 1,
        ERR, nullptr, 64, 0, SSUM, SSQ);
    bn_finish_kernel<<<1, 64, 0, stream>>>(SSUM, SSQ, (float)C, P + q.err_g,
                                           P + q.err_beta, SBN, TBN);
    zero(GMAX, 32 * 64 * 4); zero(DEN, 32 * 64 * 4);
    seg_bnmax_kernel<<<gC64, 256, 0, stream>>>(C, ERR, SBN, TBN, GIDX, GMAX);
    seg_exp_kernel<<<gC64, 256, 0, stream>>>(C, ERR, GIDX, GMAX, DEN);
    seg_div_kernel<<<gC64, 256, 0, stream>>>(C, ERR, GIDX, DEN);

    // ee_v = mlp(e_var) (pre-BN, f16); fold BN into var_nn W1d + bias
    zero(SSUM, 256); zero(SSQ, 256);
    smallin_kernel<1, false, true><<<gEsm, 256, 0, stream>>>(
        E, EFV, nullptr, P + q.vee_l1w, P + q.vee_l1b, WtVEE2[l], P + q.vee_l2b,
        1, nullptr, EE, 64, 0, SSUM, SSQ);
    bn_finish_kernel<<<1, 64, 0, stream>>>(SSUM, SSQ, (float)E, P + q.vee_g,
                                           P + q.vee_beta, SBN, TBN);
    fold_ee_kernel<<<1, 64, 0, stream>>>(P + q.vnn_l1w, 129, SBN, TBN,
                                         P + q.vnn_l1b, WtD, BIAS1);

    // var messages -> aggregate into constraints
    zero(AGG, (size_t)C * 64 * 4); zero(SSUM, 256); zero(SSQ, 256);
    edge_msg_kernel<<<gMsg, 128, 0, stream>>>(
        E, tgt_c, xc_cur, 64, 0, WtVA[l], src_v, XV, 192, colv, WtVB[l],
        nullptr, nullptr, nullptr, src_v, AV, P + q.vnn_l1w + 128 * 64, EE, WtD,
        BIAS1, WtV2[l], P + q.vnn_l2b, tgt_c, AGG, SSUM, SSQ);
    bn_finish_kernel<<<1, 64, 0, stream>>>(SSUM, SSQ, (float)E, P + q.vnn_g,
                                           P + q.vnn_beta, SBN, TBN);
    agg_finish_kernel<<<gC64, 256, 0, stream>>>(C, AGG, CNTC, SBN, TBN, xc_new,
                                                64, 0);

    // ee_c = mlp(e_con); fold BN into con_nn W1d + bias
    zero(SSUM, 256); zero(SSQ, 256);
    smallin_kernel<1, false, true><<<gEsm, 256, 0, stream>>>(
        E, EFC, nullptr, P + q.cee_l1w, P + q.cee_l1b, WtCEE2[l], P + q.cee_l2b,
        1, nullptr, EE, 64, 0, SSUM, SSQ);
    bn_finish_kernel<<<1, 64, 0, stream>>>(SSUM, SSQ, (float)E, P + q.cee_g,
                                           P + q.cee_beta, SBN, TBN);
    fold_ee_kernel<<<1, 64, 0, stream>>>(P + q.cnn_l1w, 192, SBN, TBN,
                                         P + q.cnn_l1b, WtD, BIAS1);

    // con messages -> aggregate into variables (new x_var column)
    zero(AGG, (size_t)V * 64 * 4); zero(SSUM, 256); zero(SSQ, 256);
    edge_msg_kernel<<<gMsg, 128, 0, stream>>>(
        E, tgt_v, XV, 192, colv, WtCA[l], src_c, xc_new, 64, 0, WtCB[l], src_c,
        ERR, WtCC[l], nullptr, nullptr, nullptr, EE, WtD, BIAS1, WtC2[l],
        P + q.cnn_l2b, tgt_v, AGG, SSUM, SSQ);
    bn_finish_kernel<<<1, 64, 0, stream>>>(SSUM, SSQ, (float)E, P + q.cnn_g,
                                           P + q.cnn_beta, SBN, TBN);
    agg_finish_kernel<<<gV64, 256, 0, stream>>>(V, AGG, CNTV, SBN, TBN, XV, 192,
                                                64 * (l + 1));
  }

  // ---- head: lin1(192->64) relu, lin2, lin3, lin4(64->1) ----
  mm_rows_kernel<<<gV, 256, 0, stream>>>(V, XV, 192, 0, 192, WtL1, P + l1b, 1,
                                         H1, 64, 0);
  mm_rows_kernel<<<gV, 256, 0, stream>>>(V, H1, 64, 0, 64, WtL2, P + l2b, 1,
                                         H2, 64, 0);
  mm_rows_kernel<<<gV, 256, 0, stream>>>(V, H2, 64, 0, 64, WtL3, P + l3b, 1,
                                         H1, 64, 0);
  dot_head_kernel<<<(V + 255) / 256, 256, 0, stream>>>(V, H1, P + l4w, P + l4b,
                                                       0, (float*)d_out);
  (void)n_in; (void)out_size; (void)ws_size;
}
